// GraphClassificationModel_9672266351386
// MI455X (gfx1250) — matrix operand order repaired and checked by reading, not compiled, not run on hardware
//
#include <hip/hip_runtime.h>
#include <hip/hip_bf16.h>

#define NN 30000
#define EE 240000
#define NB 60
#define NSEG 61            // B + 1 segments (last = dummy slots)
#define NLBL 10
#define GEPS 1e-5f
#define HASH_SZ 524288u
#define HASH_MASK (HASH_SZ - 1u)
#define MATCH_ITERS 80
#define TB 256

typedef __attribute__((ext_vector_type(16))) __bf16 v16bf;
typedef __attribute__((ext_vector_type(2)))  __bf16 bf16x2;
typedef __attribute__((ext_vector_type(8)))  float  v8f;
typedef unsigned long long u64;
typedef unsigned int u32;

static __device__ __forceinline__ u32 f2bfbits(float f) {
  u32 u = __float_as_uint(f);
  return (u + 0x7FFFu + ((u >> 16) & 1u)) >> 16;   // round-to-nearest-even
}
static __device__ __forceinline__ __bf16 f2bf(float f) {
  unsigned short s = (unsigned short)f2bfbits(f);
  return __builtin_bit_cast(__bf16, s);
}

static __device__ __forceinline__ void atomicMaxF(float* addr, float val) {
  if (val >= 0.0f) atomicMax((int*)addr, __float_as_int(val));
  else             atomicMin((u32*)addr, __float_as_uint(val));
}

// ---------------- generic fills / copies ----------------
__global__ void k_fillF(float* p, float v, int n) {
  int i = blockIdx.x * blockDim.x + threadIdx.x; if (i < n) p[i] = v;
}
__global__ void k_fillI(int* p, int v, int n) {
  int i = blockIdx.x * blockDim.x + threadIdx.x; if (i < n) p[i] = v;
}
__global__ void k_fillU64(u64* p, u64 v, int n) {
  int i = blockIdx.x * blockDim.x + threadIdx.x; if (i < n) p[i] = v;
}
__global__ void k_copyF(const float* a, float* b, int n) {
  int i = blockIdx.x * blockDim.x + threadIdx.x; if (i < n) b[i] = a[i];
}
__global__ void k_copyI(const int* a, int* b, int n) {
  int i = blockIdx.x * blockDim.x + threadIdx.x; if (i < n) b[i] = a[i];
}
__global__ void k_iota(int* p, int n) {
  int i = blockIdx.x * blockDim.x + threadIdx.x; if (i < n) p[i] = i;
}
__global__ void k_initedges(const int* e, int* src, int* dst, int* em) {
  int i = blockIdx.x * blockDim.x + threadIdx.x;
  if (i < EE) { src[i] = e[i]; dst[i] = e[EE + i]; em[i] = 1; }
}

// ---------------- WMMA GEMM: C[MxNc] = A[MxK](f32) * Bw[KxNc](bf16) ----------------
// Block: 256 threads = 8 waves; block tile 128(M) x 64(N); wave tile 16(M) x 64(N).
// K-step 32: A staged f32->bf16 into LDS [128][32]; B staged transposed [64][32]
// so each lane's K-pairs are contiguous bf16x2 in LDS.
__global__ __launch_bounds__(256) void k_gemm_wmma(
    const float* __restrict__ A, const __bf16* __restrict__ Bw,
    float* __restrict__ C, int M, int K, int Nc) {
  __shared__ __align__(16) __bf16 lA[128 * 32];
  __shared__ __align__(16) __bf16 lB[64 * 32];   // [col][k]
  int nblk = Nc >> 6;
  int bm = blockIdx.x / nblk, bn = blockIdx.x % nblk;
  int t = threadIdx.x;
  int wave = t >> 5, lane = t & 31;
  int r = lane & 15, half = lane >> 4;
  // A staging role: thread t covers row arow, 16 consecutive k
  int arow = t >> 1;
  int acol = (t & 1) << 4;
  int garow = bm * 128 + arow; if (garow >= M) garow = M - 1;
  // B staging role: thread t covers k-row brow, 8 consecutive cols
  int brow = t >> 3;
  int bcol = (t & 7) << 3;
  v8f acc0 = {}, acc1 = {}, acc2 = {}, acc3 = {};
  for (int k0 = 0; k0 < K; k0 += 32) {
    // ---- stage A (f32 -> bf16, packed u32 LDS stores) ----
    const float* ap = A + (size_t)garow * K + k0 + acol;
    float4 f0 = *(const float4*)(ap + 0);
    float4 f1 = *(const float4*)(ap + 4);
    float4 f2 = *(const float4*)(ap + 8);
    float4 f3 = *(const float4*)(ap + 12);
    float fv[16] = {f0.x, f0.y, f0.z, f0.w, f1.x, f1.y, f1.z, f1.w,
                    f2.x, f2.y, f2.z, f2.w, f3.x, f3.y, f3.z, f3.w};
    u32* lap = (u32*)&lA[arow * 32 + acol];
#pragma unroll
    for (int i = 0; i < 8; ++i)
      lap[i] = f2bfbits(fv[2 * i]) | (f2bfbits(fv[2 * i + 1]) << 16);
    // ---- stage B (coalesced b128 global load, transposed LDS store) ----
    const __bf16* bp = Bw + (size_t)(k0 + brow) * Nc + bn * 64 + bcol;
    uint4 braw = *(const uint4*)bp;
    u32 bw[4] = {braw.x, braw.y, braw.z, braw.w};
#pragma unroll
    for (int i = 0; i < 4; ++i) {
      lB[(bcol + 2 * i    ) * 32 + brow] = __builtin_bit_cast(__bf16, (unsigned short)(bw[i] & 0xFFFFu));
      lB[(bcol + 2 * i + 1) * 32 + brow] = __builtin_bit_cast(__bf16, (unsigned short)(bw[i] >> 16));
    }
    __syncthreads();
    // ---- A fragment for this wave's 16-row strip ----
    v16bf a;
#pragma unroll
    for (int v = 0; v < 8; ++v) {
      int ka = ((v >> 2) << 4) + (half << 3) + ((v & 3) << 1);
      bf16x2 p = *(const bf16x2*)&lA[(wave * 16 + r) * 32 + ka];
      a[2 * v] = p.x; a[2 * v + 1] = p.y;
    }
    // ---- 4 B fragments + 4 WMMAs (A reused) ----
#pragma unroll
    for (int j = 0; j < 4; ++j) {
      v16bf b;
#pragma unroll
      for (int v = 0; v < 8; ++v) {
        int kb = ((v >> 2) << 4) + (half << 3) + ((v & 3) << 1);
        bf16x2 p = *(const bf16x2*)&lB[(j * 16 + r) * 32 + kb];
        b[2 * v] = p.x; b[2 * v + 1] = p.y;
      }
      if (j == 0) acc0 = __builtin_amdgcn_wmma_f32_16x16x32_bf16(false, a, false, b, (short)0, acc0, false, false);
      if (j == 1) acc1 = __builtin_amdgcn_wmma_f32_16x16x32_bf16(false, a, false, b, (short)0, acc1, false, false);
      if (j == 2) acc2 = __builtin_amdgcn_wmma_f32_16x16x32_bf16(false, a, false, b, (short)0, acc2, false, false);
      if (j == 3) acc3 = __builtin_amdgcn_wmma_f32_16x16x32_bf16(false, a, false, b, (short)0, acc3, false, false);
    }
    __syncthreads();
  }
  // ---- store D tiles (lanes 0-15 -> M=v, lanes 16-31 -> M=8+v) ----
  int mbase = bm * 128 + wave * 16 + (half << 3);
#pragma unroll
  for (int v = 0; v < 8; ++v) {
    int mrow = mbase + v;
    if (mrow >= M) continue;
    float* Crow = C + (size_t)mrow * Nc + bn * 64 + r;
    Crow[0]  = acc0[v];
    Crow[16] = acc1[v];
    Crow[32] = acc2[v];
    Crow[48] = acc3[v];
  }
}

__global__ void k_cvtw(const float* W, __bf16* Wb, int n) {
  int i = blockIdx.x * blockDim.x + threadIdx.x; if (i < n) Wb[i] = f2bf(W[i]);
}

// ---------------- GCN ----------------
__global__ void k_degsct(const int* dst, const int* em, float* deg) {
  int e = blockIdx.x * blockDim.x + threadIdx.x;
  if (e < EE && em[e]) atomicAdd(&deg[dst[e]], 1.0f);
}
__global__ void k_rsqrt(const float* deg, float* dis, int n) {
  int i = blockIdx.x * blockDim.x + threadIdx.x; if (i < n) dis[i] = rsqrtf(deg[i]);
}
__global__ void k_aggself(const float* h, const float* dis, float* agg, int C, int total) {
  int i = blockIdx.x * blockDim.x + threadIdx.x; if (i >= total) return;
  int n = i / C; agg[i] = h[i] * dis[n] * dis[n];
}
__global__ void k_aggsct(const float* h, const float* dis, const int* src,
                         const int* dst, const int* em, float* agg, int C) {
  int i = blockIdx.x * blockDim.x + threadIdx.x;
  int e = i / C, c = i - e * C;
  if (e >= EE || !em[e]) return;
  int s = src[e], d = dst[e];
  atomicAdd(&agg[(size_t)d * C + c], h[(size_t)s * C + c] * dis[s] * dis[d]);
}
__global__ void k_biasrelu(float* x, const float* bias, int C, int total) {
  int i = blockIdx.x * blockDim.x + threadIdx.x; if (i >= total) return;
  x[i] = fmaxf(x[i] + bias[i % C], 0.0f);
}

// ---------------- GraphNorm ----------------
__global__ void k_segcnt(const int* bt, float* cnt) {
  int n = blockIdx.x * blockDim.x + threadIdx.x;
  if (n < NN) atomicAdd(&cnt[bt[n]], 1.0f);
}
__global__ void k_segsum(const float* x, const int* bt, float* ssum, int C, int total) {
  int i = blockIdx.x * blockDim.x + threadIdx.x; if (i >= total) return;
  int n = i / C, c = i - n * C;
  atomicAdd(&ssum[(size_t)bt[n] * C + c], x[i]);
}
__global__ void k_center(float* x, const int* bt, const float* ssum, const float* cnt,
                         const float* alpha, float* ssq, int C, int total) {
  int i = blockIdx.x * blockDim.x + threadIdx.x; if (i >= total) return;
  int n = i / C, c = i - n * C, b = bt[n];
  float m = ssum[(size_t)b * C + c] / fmaxf(cnt[b], 1.0f);
  float o = x[i] - alpha[c] * m;
  x[i] = o;
  atomicAdd(&ssq[(size_t)b * C + c], o * o);
}
__global__ void k_gnfin(float* x, const int* bt, const float* ssq, const float* cnt,
                        const float* gam, const float* bet, int C, int total) {
  int i = blockIdx.x * blockDim.x + threadIdx.x; if (i >= total) return;
  int n = i / C, c = i - n * C, b = bt[n];
  float var = ssq[(size_t)b * C + c] / fmaxf(cnt[b], 1.0f);
  x[i] = gam[c] * x[i] * rsqrtf(var + GEPS) + bet[c];
}

// ---------------- EdgePooling ----------------
__global__ void k_proj(const float* x, const float* w, int C, float* p1, float* p2) {
  int n = blockIdx.x * blockDim.x + threadIdx.x; if (n >= NN) return;
  float s1 = 0.f, s2 = 0.f;
  for (int c = 0; c < C; ++c) { float xv = x[(size_t)n * C + c]; s1 += xv * w[c]; s2 += xv * w[C + c]; }
  p1[n] = s1; p2[n] = s2;
}
__global__ void k_raw(const float* p1, const float* p2, const int* src, const int* dst,
                      const float* bias, float* raw) {
  int e = blockIdx.x * blockDim.x + threadIdx.x; if (e >= EE) return;
  raw[e] = p1[src[e]] + p2[dst[e]] + bias[0];
}
__global__ void k_segmax(const float* raw, const int* dst, const int* em, float* mx) {
  int e = blockIdx.x * blockDim.x + threadIdx.x;
  if (e < EE && em[e]) atomicMaxF(&mx[dst[e]], raw[e]);
}
__global__ void k_fixmx(float* mx, int n) {
  int i = blockIdx.x * blockDim.x + threadIdx.x;
  if (i < n) mx[i] = (mx[i] > -1e29f) ? mx[i] : 0.0f;
}
__global__ void k_exp(const float* raw, const float* mx, const int* dst, const int* em,
                      float* evals, float* den) {
  int e = blockIdx.x * blockDim.x + threadIdx.x; if (e >= EE) return;
  float ev = em[e] ? expf(raw[e] - mx[dst[e]]) : 0.0f;
  evals[e] = ev;
  if (em[e]) atomicAdd(&den[dst[e]], ev);
}
__global__ void k_scorekey(const float* evals, const float* den, const int* dst,
                           const int* em, float* score, u64* key) {
  int e = blockIdx.x * blockDim.x + threadIdx.x; if (e >= EE) return;
  float s = evals[e] / fmaxf(den[dst[e]], 1e-30f) + 0.5f;
  score[e] = s;
  u32 u = __float_as_uint(s);
  u32 ord  = (u & 0x80000000u) ? ~u : (u | 0x80000000u);  // monotone ascending
  u32 desc = ~ord;                                         // descending score
  key[e] = em[e] ? (((u64)desc << 32) | (u32)e) : ~0ull;
}
__global__ void k_minkey(const u64* key, const int* src, const int* dst, const int* em,
                         const int* matched, const int* chosen, u64* mk) {
  int e = blockIdx.x * blockDim.x + threadIdx.x; if (e >= EE) return;
  int s = src[e], d = dst[e];
  if (!em[e] || chosen[e] || matched[s] || matched[d]) return;
  u64 k = key[e];
  atomicMin(&mk[s], k); atomicMin(&mk[d], k);
}
__global__ void k_select(const u64* key, const int* src, const int* dst, const int* em,
                         int* matched, int* chosen, const u64* mk) {
  int e = blockIdx.x * blockDim.x + threadIdx.x; if (e >= EE) return;
  if (!em[e] || chosen[e]) return;
  int s = src[e], d = dst[e];
  if (matched[s] || matched[d]) return;
  u64 k = key[e];
  if (k == mk[s] && k == mk[d]) {
    chosen[e] = 1;
    atomicExch(&matched[s], 1);
    atomicExch(&matched[d], 1);
  }
}
__global__ void k_applych(const int* chosen, const int* src, const int* dst,
                          const float* score, int* cluster, float* nscore) {
  int e = blockIdx.x * blockDim.x + threadIdx.x; if (e >= EE) return;
  if (chosen[e]) { cluster[dst[e]] = src[e]; nscore[src[e]] = score[e]; }
}
__global__ void k_newxsct(const float* x, const int* cluster, float* nx, int C, int total) {
  int i = blockIdx.x * blockDim.x + threadIdx.x; if (i >= total) return;
  int n = i / C, c = i - n * C;
  atomicAdd(&nx[(size_t)cluster[n] * C + c], x[i]);
}
__global__ void k_newxscale(float* nx, const float* nscore, int C, int total) {
  int i = blockIdx.x * blockDim.x + threadIdx.x; if (i >= total) return;
  nx[i] *= nscore[i / C];
}
__global__ void k_newbatch(const int* cluster, const int* bt, int* btn) {
  int n = blockIdx.x * blockDim.x + threadIdx.x; if (n >= NN) return;
  btn[cluster[n]] = bt[n];
}
__global__ void k_remapkey(const int* cluster, const int* src, const int* dst,
                           const int* em, u32* keys) {
  int e = blockIdx.x * blockDim.x + threadIdx.x; if (e >= EE) return;
  keys[e] = em[e] ? ((u32)cluster[src[e]] * (u32)NN + (u32)cluster[dst[e]]) : 0xFFFFFFFFu;
}
__global__ void k_hashins(const u32* keys, u64* table) {
  int e = blockIdx.x * blockDim.x + threadIdx.x; if (e >= EE) return;
  u32 k = keys[e]; if (k == 0xFFFFFFFFu) return;
  u32 pos = (k * 2654435761u) & HASH_MASK;
  u64 desired = ((u64)k << 32) | (u32)e;
  for (;;) {
    u64 cur = table[pos];
    if (cur == ~0ull) {
      u64 prev = atomicCAS(&table[pos], ~0ull, desired);
      if (prev == ~0ull) return;
      cur = prev;
    }
    if ((u32)(cur >> 32) == k) { atomicMin(&table[pos], desired); return; }
    pos = (pos + 1u) & HASH_MASK;
  }
}
__global__ void k_remapapply(const u32* keys, const u64* table, int* src, int* dst, int* em) {
  int e = blockIdx.x * blockDim.x + threadIdx.x; if (e >= EE) return;
  u32 k = keys[e];
  if (k == 0xFFFFFFFFu) { em[e] = 0; src[e] = 0; dst[e] = 0; return; }
  u32 pos = (k * 2654435761u) & HASH_MASK;
  for (;;) {
    u64 cur = table[pos];
    if ((u32)(cur >> 32) == k) {
      if ((u32)cur == (u32)e) { src[e] = (int)(k / (u32)NN); dst[e] = (int)(k % (u32)NN); em[e] = 1; }
      else { em[e] = 0; src[e] = 0; dst[e] = 0; }
      return;
    }
    pos = (pos + 1u) & HASH_MASK;
  }
}

// ---------------- Readout head ----------------
__global__ void k_poolacc(const float* x, const int* bt, float* psum, float* pmax, int total) {
  int i = blockIdx.x * blockDim.x + threadIdx.x; if (i >= total) return;
  int n = i >> 9, c = i & 511;
  int b = bt[n]; if (b >= NB) return;
  float xv = x[i];
  atomicAdd(&psum[(size_t)b * 512 + c], xv);
  atomicMaxF(&pmax[(size_t)b * 512 + c], xv);
}
__global__ void k_head(float* psum, const float* pmax, const float* cnt, int total) {
  int i = blockIdx.x * blockDim.x + threadIdx.x; if (i >= total) return;
  int b = i >> 9;
  psum[i] = psum[i] / fmaxf(cnt[b], 1.0f) + pmax[i];
}
__global__ void k_lin(const float* g, const float* linW, const float* linb,
                      const float* bnG, const float* bnB, float* h1) {
  int i = blockIdx.x * blockDim.x + threadIdx.x; if (i >= NB * 1024) return;
  int b = i >> 10, j = i & 1023;
  float acc = linb[j];
  for (int k = 0; k < 512; ++k) acc += g[b * 512 + k] * linW[k * 1024 + j];
  h1[i] = acc * (bnG[j] * rsqrtf(1.0f + GEPS)) + bnB[j];
}
__global__ void k_cls(const float* h1, const float* clsW, const float* clsb, float* out) {
  int i = blockIdx.x * blockDim.x + threadIdx.x; if (i >= NB * NLBL) return;
  int b = i / NLBL, j = i - b * NLBL;
  float acc = clsb[j];
  for (int k = 0; k < 1024; ++k) acc += h1[b * 1024 + k] * clsW[k * NLBL + j];
  out[i] = acc;
}
__global__ void k_softmax(const float* logits, float* probs) {
  int b = blockIdx.x * blockDim.x + threadIdx.x; if (b >= NB) return;
  float m = -1e30f;
  for (int j = 0; j < NLBL; ++j) m = fmaxf(m, logits[b * NLBL + j]);
  float s = 0.f, e[NLBL];
  for (int j = 0; j < NLBL; ++j) { e[j] = expf(logits[b * NLBL + j] - m); s += e[j]; }
  for (int j = 0; j < NLBL; ++j) probs[b * NLBL + j] = e[j] / s;
}

static inline int gsz(long long n, int b) { return (int)((n + b - 1) / b); }

extern "C" void kernel_launch(void* const* d_in, const int* in_sizes, int n_in,
                              void* d_out, int out_size, void* d_ws, size_t ws_size,
                              hipStream_t stream) {
  (void)in_sizes; (void)n_in; (void)out_size; (void)ws_size;
  // -------- workspace layout --------
  char* w = (char*)d_ws;
  auto take = [&](size_t bytes) -> void* {
    void* p = (void*)w; w += (bytes + 255) & ~(size_t)255; return p;
  };
  float* bufP   = (float*)take((size_t)NN * 512 * 4);
  float* bufQ   = (float*)take((size_t)NN * 512 * 4);
  __bf16* wbf   = (__bf16*)take((size_t)256 * 512 * 2);
  float* deg    = (float*)take((size_t)NN * 4);
  float* dis    = (float*)take((size_t)NN * 4);
  int*   btA    = (int*)take((size_t)NN * 4);
  int*   btB    = (int*)take((size_t)NN * 4);
  int*   esrc   = (int*)take((size_t)EE * 4);
  int*   edst   = (int*)take((size_t)EE * 4);
  int*   eem    = (int*)take((size_t)EE * 4);
  u64*   key64  = (u64*)take((size_t)EE * 8);
  float* raw    = (float*)take((size_t)EE * 4);
  float* evals  = (float*)take((size_t)EE * 4);
  float* score  = (float*)take((size_t)EE * 4);
  int*   chosen = (int*)take((size_t)EE * 4);
  float* p1     = (float*)take((size_t)NN * 4);
  float* p2     = (float*)take((size_t)NN * 4);
  float* mx     = (float*)take((size_t)NN * 4);
  float* den    = (float*)take((size_t)NN * 4);
  u64*   minkey = (u64*)take((size_t)NN * 8);
  int*   matched= (int*)take((size_t)NN * 4);
  int*   cluster= (int*)take((size_t)NN * 4);
  float* nscore = (float*)take((size_t)NN * 4);
  float* segcnt = (float*)take((size_t)NSEG * 4);
  float* segsum = (float*)take((size_t)NSEG * 512 * 4);
  float* segsq  = (float*)take((size_t)NSEG * 512 * 4);
  u64*   hasht  = (u64*)take((size_t)HASH_SZ * 8);
  float* psum   = (float*)take((size_t)NB * 512 * 4);
  float* pmax   = (float*)take((size_t)NB * 512 * 4);
  float* h1     = (float*)take((size_t)NB * 1024 * 4);

  // -------- inputs --------
  const float* vin  = (const float*)d_in[0];
  const int*   ein  = (const int*)d_in[1];
  const int*   btin = (const int*)d_in[2];
  const int Cins[3]  = {64, 128, 256};
  const int Couts[3] = {128, 256, 512};
  const int iW[3]   = {3, 10, 17}, ibi[3] = {4, 11, 18}, ial[3] = {5, 12, 19};
  const int igm[3]  = {6, 13, 20}, ibe[3] = {7, 14, 21};
  const int iepw[3] = {8, 15, 22}, iepb[3] = {9, 16, 23};

  k_copyF<<<gsz((long long)NN * 64, TB), TB, 0, stream>>>(vin, bufP, NN * 64);
  k_initedges<<<gsz(EE, TB), TB, 0, stream>>>(ein, esrc, edst, eem);
  k_copyI<<<gsz(NN, TB), TB, 0, stream>>>(btin, btA, NN);

  float* cur = bufP; float* oth = bufQ;
  int* bt = btA; int* btN = btB;

  for (int s = 0; s < 3; ++s) {
    int K = Cins[s], C = Couts[s];
    const float* W    = (const float*)d_in[iW[s]];
    const float* bias = (const float*)d_in[ibi[s]];
    const float* alph = (const float*)d_in[ial[s]];
    const float* gam  = (const float*)d_in[igm[s]];
    const float* bet  = (const float*)d_in[ibe[s]];
    const float* epw  = (const float*)d_in[iepw[s]];
    const float* epb  = (const float*)d_in[iepb[s]];
    long long nc = (long long)NN * C;
    long long ec = (long long)EE * C;

    // ---- GCN: h = x @ W  (bf16 WMMA, LDS-staged) ----
    k_cvtw<<<gsz((long long)K * C, TB), TB, 0, stream>>>(W, wbf, K * C);
    int gx = ((NN + 127) / 128) * (C / 64);
    k_gemm_wmma<<<gx, 256, 0, stream>>>(cur, wbf, oth, NN, K, C);
    // degrees / normalization
    k_fillF<<<gsz(NN, TB), TB, 0, stream>>>(deg, 1.0f, NN);
    k_degsct<<<gsz(EE, TB), TB, 0, stream>>>(edst, eem, deg);
    k_rsqrt<<<gsz(NN, TB), TB, 0, stream>>>(deg, dis, NN);
    // agg = h*dis^2 + scatter(h[src]*dis[src]*dis[dst]) ; write into cur (x is dead)
    k_aggself<<<gsz(nc, TB), TB, 0, stream>>>(oth, dis, cur, C, (int)nc);
    k_aggsct<<<gsz(ec, TB), TB, 0, stream>>>(oth, dis, esrc, edst, eem, cur, C);
    k_biasrelu<<<gsz(nc, TB), TB, 0, stream>>>(cur, bias, C, (int)nc);

    // ---- GraphNorm (in place on cur) ----
    k_fillF<<<gsz(NSEG, TB), TB, 0, stream>>>(segcnt, 0.f, NSEG);
    k_segcnt<<<gsz(NN, TB), TB, 0, stream>>>(bt, segcnt);
    k_fillF<<<gsz((long long)NSEG * C, TB), TB, 0, stream>>>(segsum, 0.f, NSEG * C);
    k_segsum<<<gsz(nc, TB), TB, 0, stream>>>(cur, bt, segsum, C, (int)nc);
    k_fillF<<<gsz((long long)NSEG * C, TB), TB, 0, stream>>>(segsq, 0.f, NSEG * C);
    k_center<<<gsz(nc, TB), TB, 0, stream>>>(cur, bt, segsum, segcnt, alph, segsq, C, (int)nc);
    k_gnfin<<<gsz(nc, TB), TB, 0, stream>>>(cur, bt, segsq, segcnt, gam, bet, C, (int)nc);

    // ---- EdgePooling ----
    k_proj<<<gsz(NN, TB), TB, 0, stream>>>(cur, epw, C, p1, p2);
    k_raw<<<gsz(EE, TB), TB, 0, stream>>>(p1, p2, esrc, edst, epb, raw);
    k_fillF<<<gsz(NN, TB), TB, 0, stream>>>(mx, -1e30f, NN);
    k_segmax<<<gsz(EE, TB), TB, 0, stream>>>(raw, edst, eem, mx);
    k_fixmx<<<gsz(NN, TB), TB, 0, stream>>>(mx, NN);
    k_fillF<<<gsz(NN, TB), TB, 0, stream>>>(den, 0.f, NN);
    k_exp<<<gsz(EE, TB), TB, 0, stream>>>(raw, mx, edst, eem, evals, den);
    k_scorekey<<<gsz(EE, TB), TB, 0, stream>>>(evals, den, edst, eem, score, key64);
    k_fillI<<<gsz(EE, TB), TB, 0, stream>>>(chosen, 0, EE);
    k_fillI<<<gsz(NN, TB), TB, 0, stream>>>(matched, 0, NN);
    for (int it = 0; it < MATCH_ITERS; ++it) {
      k_fillU64<<<gsz(NN, TB), TB, 0, stream>>>(minkey, ~0ull, NN);
      k_minkey<<<gsz(EE, TB), TB, 0, stream>>>(key64, esrc, edst, eem, matched, chosen, minkey);
      k_select<<<gsz(EE, TB), TB, 0, stream>>>(key64, esrc, edst, eem, matched, chosen, minkey);
    }
    k_iota<<<gsz(NN, TB), TB, 0, stream>>>(cluster, NN);
    k_fillF<<<gsz(NN, TB), TB, 0, stream>>>(nscore, 1.0f, NN);
    k_fillI<<<gsz(NN, TB), TB, 0, stream>>>(btN, NB, NN);
    k_applych<<<gsz(EE, TB), TB, 0, stream>>>(chosen, esrc, edst, score, cluster, nscore);
    // new_x into oth (h is dead)
    k_fillF<<<gsz(nc, TB), TB, 0, stream>>>(oth, 0.f, (int)nc);
    k_newxsct<<<gsz(nc, TB), TB, 0, stream>>>(cur, cluster, oth, C, (int)nc);
    k_newxscale<<<gsz(nc, TB), TB, 0, stream>>>(oth, nscore, C, (int)nc);
    k_newbatch<<<gsz(NN, TB), TB, 0, stream>>>(cluster, bt, btN);
    // remap + deterministic dedup of edges
    u32* keys32 = (u32*)key64;   // key64 no longer needed
    k_remapkey<<<gsz(EE, TB), TB, 0, stream>>>(cluster, esrc, edst, eem, keys32);
    k_fillU64<<<gsz(HASH_SZ, TB), TB, 0, stream>>>(hasht, ~0ull, (int)HASH_SZ);
    k_hashins<<<gsz(EE, TB), TB, 0, stream>>>(keys32, hasht);
    k_remapapply<<<gsz(EE, TB), TB, 0, stream>>>(keys32, hasht, esrc, edst, eem);
    // swap buffers
    { float* t = cur; cur = oth; oth = t; }
    { int* t = bt; bt = btN; btN = t; }
  }

  // ---- readout: (mean + max) pooling -> linear+BN -> classifier -> softmax ----
  k_fillF<<<gsz(NSEG, TB), TB, 0, stream>>>(segcnt, 0.f, NSEG);
  k_segcnt<<<gsz(NN, TB), TB, 0, stream>>>(bt, segcnt);
  k_fillF<<<gsz((long long)NB * 512, TB), TB, 0, stream>>>(psum, 0.f, NB * 512);
  k_fillF<<<gsz((long long)NB * 512, TB), TB, 0, stream>>>(pmax, -1e30f, NB * 512);
  k_poolacc<<<gsz((long long)NN * 512, TB), TB, 0, stream>>>(cur, bt, psum, pmax, NN * 512);
  k_head<<<gsz((long long)NB * 512, TB), TB, 0, stream>>>(psum, pmax, segcnt, NB * 512);
  const float* linW = (const float*)d_in[24];
  const float* linb = (const float*)d_in[25];
  const float* bnG  = (const float*)d_in[26];
  const float* bnB  = (const float*)d_in[27];
  const float* clsW = (const float*)d_in[28];
  const float* clsb = (const float*)d_in[29];
  k_lin<<<gsz((long long)NB * 1024, TB), TB, 0, stream>>>(psum, linW, linb, bnG, bnB, h1);
  float* out = (float*)d_out;
  k_cls<<<gsz((long long)NB * NLBL, TB), TB, 0, stream>>>(h1, clsW, clsb, out);
  k_softmax<<<gsz(NB, TB), TB, 0, stream>>>(out, out + NB * NLBL);
}
